// TrustGNN_75007308857923
// MI455X (gfx1250) — matrix operand order repaired and checked by reading, not compiled, not run on hardware
//
#include <hip/hip_runtime.h>
#include <cstdint>
#include <cstddef>

#define NNODES 50000
#define NEDGES 800000
#define ETOT   (NEDGES + NNODES)

typedef __attribute__((ext_vector_type(16))) _Float16 v16h;
typedef __attribute__((ext_vector_type(8)))  _Float16 v8h;
typedef __attribute__((ext_vector_type(8)))  float    v8f;

// ---------------- conversion kernels ----------------
__global__ void k_f32_to_f16(const float* __restrict__ s, _Float16* __restrict__ d, int n) {
  int i = blockIdx.x * blockDim.x + threadIdx.x;
  if (i < n) d[i] = (_Float16)s[i];
}

// src[rows][cols] (f32, row-major)  ->  dst[cols][rows] (f16)
__global__ void k_transpose_to_f16(const float* __restrict__ s, _Float16* __restrict__ d,
                                   int rows, int cols) {
  int i = blockIdx.x * blockDim.x + threadIdx.x;
  if (i < rows * cols) {
    int r = i / cols, c = i % cols;
    d[c * rows + r] = (_Float16)s[i];
  }
}

// ---------------- WMMA GEMM ----------------
// C[M x Nc] (f32) = A[M x K] (f16 row-major) * B  where B is supplied transposed: Bt[Nc x K].
// One 16x16 output tile per wave32, K stepped by 32 (v_wmma_f32_16x16x32_f16, f32 accum).
__global__ void k_wmma_gemm(const _Float16* __restrict__ A, const _Float16* __restrict__ Bt,
                            float* __restrict__ C, int M, int K, int Nc) {
  int lane = threadIdx.x & 31;
  int wave = (blockIdx.x * blockDim.x + threadIdx.x) >> 5;
  int ntn  = Nc >> 4;
  int tiles = (M >> 4) * ntn;
  if (wave >= tiles) return;            // wave-uniform guard: EXEC all-ones at WMMA
  int m0  = (wave / ntn) << 4;
  int n0  = (wave % ntn) << 4;
  int row = lane & 15;                  // A: M index / B: N index / D: column
  int g   = lane >> 4;                  // half-wave group

  // A fragment: lane holds row m0+row, halves j -> K=k0+g*16+j  (contiguous)
  const _Float16* ap = A  + (size_t)(m0 + row) * K + g * 16;
  // B fragment (from Bt): lane holds column n0+row, halves j -> K=k0+g*16+j  (contiguous in Bt)
  const _Float16* bp = Bt + (size_t)(n0 + row) * K + g * 16;

  v8f acc = {};
  for (int k0 = 0; k0 < K; k0 += 32) {
    v8h alo = *(const v8h*)(ap + k0);
    v8h ahi = *(const v8h*)(ap + k0 + 8);
    v16h a = __builtin_shufflevector(alo, ahi, 0,1,2,3,4,5,6,7,8,9,10,11,12,13,14,15);
    v16h b = *(const v16h*)(bp + k0);
    acc = __builtin_amdgcn_wmma_f32_16x16x32_f16(false, a, false, b, (short)0, acc, false, false);
  }
  // D layout: VGPR r <-> row m0 + r + 8*g, column n0 + (lane&15)
  union { v8f v; float f[8]; } u; u.v = acc;
  float* cp = C + (size_t)(m0 + g * 8) * Nc + n0 + row;
  #pragma unroll
  for (int r = 0; r < 8; ++r) cp[(size_t)r * Nc] = u.f[r];
}

// ---------------- attention coefficients: asrc[n,h]=<h1[n,h,:],a_src[h,:]> ----------------
__global__ void k_attn_coef(const float* __restrict__ h, const float* __restrict__ a_src,
                            const float* __restrict__ a_dst, float* __restrict__ asrc,
                            float* __restrict__ adst, int n, int Hh, int Cc) {
  int i = blockIdx.x * blockDim.x + threadIdx.x;
  if (i >= n * Hh) return;
  int node = i / Hh, hh = i % Hh;
  const float* hp = h + (size_t)node * Hh * Cc + (size_t)hh * Cc;
  const float* as = a_src + hh * Cc;
  const float* ad = a_dst + hh * Cc;
  float s1 = 0.f, s2 = 0.f;
  for (int c = 0; c < Cc; ++c) { float v = hp[c]; s1 += v * as[c]; s2 += v * ad[c]; }
  asrc[i] = s1; adst[i] = s2;
}

// ---------------- edge helpers ----------------
__device__ __forceinline__ void edge_ends(const long long* __restrict__ ei, int e, int& s, int& d) {
  if (e < NEDGES) { s = (int)ei[e]; d = (int)ei[NEDGES + e]; }
  else            { s = e - NEDGES; d = e - NEDGES; }      // appended self-loops
}
__device__ __forceinline__ unsigned fenc(float f) {        // order-preserving float->uint
  unsigned u = __float_as_uint(f);
  return (u & 0x80000000u) ? ~u : (u | 0x80000000u);
}
__device__ __forceinline__ float fdec(unsigned k) {
  unsigned u = (k & 0x80000000u) ? (k ^ 0x80000000u) : ~k;
  return __uint_as_float(u);
}
__device__ __forceinline__ float lrelu(float v) { return v > 0.f ? v : 0.2f * v; }

// pass A: segment max over dst via atomicMax on monotone uint key
__global__ void k_edge_max(const long long* __restrict__ ei, const float* __restrict__ asrc,
                           const float* __restrict__ adst, unsigned* __restrict__ mkey, int Hh) {
  int e = blockIdx.x * blockDim.x + threadIdx.x;
  if (e >= ETOT) return;
  int s, d; edge_ends(ei, e, s, d);
  for (int hh = 0; hh < Hh; ++hh) {
    float sc = lrelu(asrc[s * Hh + hh] + adst[d * Hh + hh]);
    atomicMax(&mkey[d * Hh + hh], fenc(sc));
  }
}

__global__ void k_decode_max(const unsigned* __restrict__ mkey, float* __restrict__ m, int n) {
  int i = blockIdx.x * blockDim.x + threadIdx.x;
  if (i < n) m[i] = mkey[i] ? fdec(mkey[i]) : 0.f;         // untouched (-inf) -> 0, per reference
}

// pass B: ex = exp(score - m[dst]); denom[dst] += ex
__global__ void k_edge_expsum(const long long* __restrict__ ei, const float* __restrict__ asrc,
                              const float* __restrict__ adst, const float* __restrict__ m,
                              float* __restrict__ ex, float* __restrict__ denom, int Hh) {
  int i = blockIdx.x * blockDim.x + threadIdx.x;
  if (i >= ETOT * Hh) return;
  int e = i / Hh, hh = i % Hh;
  int s, d; edge_ends(ei, e, s, d);
  float sc = lrelu(asrc[s * Hh + hh] + adst[d * Hh + hh]);
  float v = __expf(sc - m[d * Hh + hh]);
  ex[i] = v;
  atomicAdd(&denom[d * Hh + hh], v);
}

// pass C: out[dst, c] += (ex/denom[dst]) * h[src, c]   (thread per edge x channel)
__global__ void k_edge_aggregate(const long long* __restrict__ ei, const float* __restrict__ h,
                                 const float* __restrict__ ex, const float* __restrict__ denom,
                                 float* __restrict__ out, int Hh, int Cc) {
  long long i = (long long)blockIdx.x * blockDim.x + threadIdx.x;
  int HC = Hh * Cc;
  if (i >= (long long)ETOT * HC) return;
  int e = (int)(i / HC), c = (int)(i % HC);
  int hh = c / Cc;
  int s, d; edge_ends(ei, e, s, d);
  float alpha = ex[(size_t)e * Hh + hh] / (denom[d * Hh + hh] + 1e-16f);
  atomicAdd(&out[(size_t)d * HC + c], alpha * h[(size_t)s * HC + c]);
}

// fused bias + ELU + f16 convert (layer-2 GEMM input)
__global__ void k_bias_elu_f16(const float* __restrict__ acc, const float* __restrict__ b,
                               _Float16* __restrict__ dst, int n, int Cc) {
  int i = blockIdx.x * blockDim.x + threadIdx.x;
  if (i < n) {
    float v = acc[i] + b[i % Cc];
    v = v > 0.f ? v : expm1f(v);
    dst[i] = (_Float16)v;
  }
}

__global__ void k_bias_out(const float* __restrict__ acc, const float* __restrict__ b,
                           float* __restrict__ out, int n, int Cc) {
  int i = blockIdx.x * blockDim.x + threadIdx.x;
  if (i < n) out[i] = acc[i] + b[i % Cc];
}

// ---------------- host launch ----------------
extern "C" void kernel_launch(void* const* d_in, const int* in_sizes, int n_in,
                              void* d_out, int out_size, void* d_ws, size_t ws_size,
                              hipStream_t stream) {
  (void)in_sizes; (void)n_in; (void)out_size; (void)ws_size;
  const float*     x      = (const float*)d_in[0];
  const long long* ei     = (const long long*)d_in[1];   // int64 [2,E]
  const float*     W1     = (const float*)d_in[2];
  const float*     a_src1 = (const float*)d_in[3];
  const float*     a_dst1 = (const float*)d_in[4];
  const float*     b1     = (const float*)d_in[5];
  const float*     W2     = (const float*)d_in[6];
  const float*     a_src2 = (const float*)d_in[7];
  const float*     a_dst2 = (const float*)d_in[8];
  const float*     b2     = (const float*)d_in[9];
  float* out = (float*)d_out;

  const int N = NNODES;
  char* ws = (char*)d_ws;
  size_t off = 0;
  auto carve = [&](size_t bytes) -> char* {
    char* p = ws + off;
    off = (off + bytes + 255) & ~(size_t)255;
    return p;
  };
  _Float16* xh    = (_Float16*)carve((size_t)N * 128 * 2);   // f16 copy of x
  _Float16* W1t   = (_Float16*)carve(128 * 256 * 2);         // W1^T f16 [256x128]
  _Float16* W2t   = (_Float16*)carve(256 * 64 * 2);          // W2^T f16 [64x256]
  float*    h1    = (float*)carve((size_t)N * 256 * 4);      // layer1 GEMM out
  _Float16* helu  = (_Float16*)carve((size_t)N * 256 * 2);   // elu(out1+b1) f16
  float*    asrc  = (float*)carve((size_t)N * 4 * 4);
  float*    adst  = (float*)carve((size_t)N * 4 * 4);
  unsigned* mkey  = (unsigned*)carve((size_t)N * 4 * 4);
  float*    m     = (float*)carve((size_t)N * 4 * 4);
  float*    den   = (float*)carve((size_t)N * 4 * 4);
  float*    ex    = (float*)carve((size_t)ETOT * 4 * 4);
  float*    out1  = (float*)carve((size_t)N * 256 * 4);      // layer1 aggregation
  // layer-2 aliases of dead layer-1 buffers
  float* h2   = h1;              // h1 dead after layer-1 aggregation
  float* out2 = (float*)xh;      // xh dead after GEMM1; N*64*4 == N*128*2 bytes

  const int B = 256;
  // --- convert ---
  k_f32_to_f16<<<(N * 128 + B - 1) / B, B, 0, stream>>>(x, xh, N * 128);
  k_transpose_to_f16<<<(128 * 256 + B - 1) / B, B, 0, stream>>>(W1, W1t, 128, 256);
  k_transpose_to_f16<<<(256 * 64 + B - 1) / B, B, 0, stream>>>(W2, W2t, 256, 64);

  // --- layer 1 ---
  int waves1 = (N / 16) * (256 / 16);
  k_wmma_gemm<<<(waves1 * 32 + B - 1) / B, B, 0, stream>>>(xh, W1t, h1, N, 128, 256);
  k_attn_coef<<<(N * 4 + B - 1) / B, B, 0, stream>>>(h1, a_src1, a_dst1, asrc, adst, N, 4, 64);
  hipMemsetAsync(mkey, 0, (size_t)N * 4 * 4, stream);
  hipMemsetAsync(den,  0, (size_t)N * 4 * 4, stream);
  hipMemsetAsync(out1, 0, (size_t)N * 256 * 4, stream);
  k_edge_max<<<(ETOT + B - 1) / B, B, 0, stream>>>(ei, asrc, adst, mkey, 4);
  k_decode_max<<<(N * 4 + B - 1) / B, B, 0, stream>>>(mkey, m, N * 4);
  k_edge_expsum<<<(ETOT * 4 + B - 1) / B, B, 0, stream>>>(ei, asrc, adst, m, ex, den, 4);
  {
    long long tot = (long long)ETOT * 256;
    k_edge_aggregate<<<(unsigned)((tot + B - 1) / B), B, 0, stream>>>(ei, h1, ex, den, out1, 4, 64);
  }
  k_bias_elu_f16<<<(N * 256 + B - 1) / B, B, 0, stream>>>(out1, b1, helu, N * 256, 256);

  // --- layer 2 ---
  int waves2 = (N / 16) * (64 / 16);
  k_wmma_gemm<<<(waves2 * 32 + B - 1) / B, B, 0, stream>>>(helu, W2t, h2, N, 256, 64);
  k_attn_coef<<<(N + B - 1) / B, B, 0, stream>>>(h2, a_src2, a_dst2, asrc, adst, N, 1, 64);
  hipMemsetAsync(mkey, 0, (size_t)N * 4, stream);
  hipMemsetAsync(den,  0, (size_t)N * 4, stream);
  hipMemsetAsync(out2, 0, (size_t)N * 64 * 4, stream);
  k_edge_max<<<(ETOT + B - 1) / B, B, 0, stream>>>(ei, asrc, adst, mkey, 1);
  k_decode_max<<<(N + B - 1) / B, B, 0, stream>>>(mkey, m, N);
  k_edge_expsum<<<(ETOT + B - 1) / B, B, 0, stream>>>(ei, asrc, adst, m, ex, den, 1);
  {
    long long tot = (long long)ETOT * 64;
    k_edge_aggregate<<<(unsigned)((tot + B - 1) / B), B, 0, stream>>>(ei, h2, ex, den, out2, 1, 64);
  }
  k_bias_out<<<(N * 64 + B - 1) / B, B, 0, stream>>>(out2, b2, out, N * 64, 64);
}
